// Deform_Conv_V1_11733850653146
// MI455X (gfx1250) — compile-verified, matrix operand order checked by reading, not run on hardware
//
#include <hip/hip_runtime.h>
#include <hip/hip_bf16.h>

typedef __attribute__((ext_vector_type(16))) _Float16 v16h;
typedef __attribute__((ext_vector_type(8)))  _Float16 v8h;
typedef __attribute__((ext_vector_type(8)))  float    v8f;

#define B_    8
#define CIN   32
#define H_    160
#define W_    160
#define COUT  32
#define KSZ   3
#define KK    9
#define PAD_  1
#define HW    (H_*W_)
#define OFFC  (2*KK)   /* 18 offset channels */
#define WAVES 4        /* waves per block */

// ---------------------------------------------------------------------------
// Fully fused deformable conv:
//  Phase A: offsets = conv3x3(x, w_off)+b_off via WMMA (M=18 pad 32, K=288)
//           -> wave-private LDS transpose (18 ch x 16 px)
//  Phase B: bilinear-sample 288 x 16 tile (f16) -> WMMA einsum (M=32, K=288)
//           -> ReLU -> store.
// One wave owns one 16-pixel tile; all GEMM FLOPs on v_wmma_f32_16x16x32_f16.
// ---------------------------------------------------------------------------
__global__ __launch_bounds__(WAVES * 32) void dcn_fused_wmma_kernel(
    const float* __restrict__ x, const float* __restrict__ w_off,
    const float* __restrict__ b_off, const float* __restrict__ w_dcn,
    float* __restrict__ out)
{
  // A operands, fragment-ordered: [cout][kk*32 + cin] f16 (rows 576B, 16B aligned)
  __shared__ __align__(16) _Float16 sWoff[32][KK * CIN];   // rows 18..31 zero
  __shared__ __align__(16) _Float16 sWdcn[COUT][KK * CIN];
  // B staging, wave-private: [pixel 0..15][cin 0..31] f16 (rows 64B)
  __shared__ __align__(16) _Float16 sB[WAVES][16][CIN];
  // offset transpose, wave-private: [channel 0..17][pixel 0..15] f32
  __shared__ __align__(16) float sOff[WAVES][OFFC][16];

  for (int t = threadIdx.x; t < 32 * KK * CIN; t += blockDim.x) {
    const int co = t / (KK * CIN), r = t % (KK * CIN);
    const int kk = r / CIN, cin = r % CIN;
    // both weights are [C][32][3][3]; swizzle to [C][kk*32+cin]
    sWoff[co][kk * CIN + cin] =
        (co < OFFC) ? (_Float16)w_off[(co * CIN + cin) * KK + kk] : (_Float16)0.0f;
    sWdcn[co][kk * CIN + cin] = (_Float16)w_dcn[(co * CIN + cin) * KK + kk];
  }
  __syncthreads();

  const int wave = threadIdx.x >> 5;
  const int lane = threadIdx.x & 31;
  const int p    = lane & 15;   // fragment column (N) / A-row select
  const int hsel = lane >> 4;   // K-half select per 16-bit WMMA VGPR layout
  const int kbase = hsel * 8;

  const int tile = blockIdx.x * WAVES + wave;
  const int g    = tile * 16 + p;                 // flat pixel for this lane
  const int b = g / HW, rem = g % HW;
  const int oy = rem / W_, ox = rem % W_;

  const float* xb  = x + (size_t)b * CIN * HW;
  _Float16*    myB = &sB[wave][0][0];
  float*       myO = &sOff[wave][0][0];

  // ------------------- Phase A: offset conv via WMMA -------------------
  v8f acc0, acc1;
#pragma unroll
  for (int v = 0; v < 8; ++v) {                   // fold bias into C init
    const int r0 = v + 8 * hsel;                  // rows 0..15 (all valid)
    const int r1 = 16 + r0;                       // rows 16..31 (18..31 unused)
    acc0[v] = b_off[r0];
    acc1[v] = (r1 < OFFC) ? b_off[r1] : 0.0f;
  }

  for (int kk = 0; kk < KK; ++kk) {
    const int ky = kk / 3, kx = kk % 3;
    const int  iy = oy - PAD_ + ky, ix = ox - PAD_ + kx;
    const bool v  = (iy >= 0) & (iy < H_) & (ix >= 0) & (ix < W_);
    const int  ipix = (v ? iy : 0) * W_ + (v ? ix : 0);
    const float msk = (float)v;
#pragma unroll
    for (int c = 0; c < 16; ++c) {                // im2col column, zero-padded
      const int cin = hsel * 16 + c;
      myB[p * CIN + cin] = (_Float16)(msk * xb[cin * HW + ipix]);
    }
    asm volatile("s_wait_dscnt 0" ::: "memory");
    __builtin_amdgcn_wave_barrier();

    const v8h blo = *(const v8h*)&myB[p * CIN + kbase];
    const v8h bhi = *(const v8h*)&myB[p * CIN + kbase + 16];
    const v16h bf = __builtin_shufflevector(blo, bhi,
        0,1,2,3,4,5,6,7,8,9,10,11,12,13,14,15);

    const _Float16* wr0 = &sWoff[p     ][kk * CIN + kbase];
    const _Float16* wr1 = &sWoff[p + 16][kk * CIN + kbase];
    const v8h a0lo = *(const v8h*)(wr0);  const v8h a0hi = *(const v8h*)(wr0 + 16);
    const v8h a1lo = *(const v8h*)(wr1);  const v8h a1hi = *(const v8h*)(wr1 + 16);
    const v16h af0 = __builtin_shufflevector(a0lo, a0hi,
        0,1,2,3,4,5,6,7,8,9,10,11,12,13,14,15);
    const v16h af1 = __builtin_shufflevector(a1lo, a1hi,
        0,1,2,3,4,5,6,7,8,9,10,11,12,13,14,15);

    acc0 = __builtin_amdgcn_wmma_f32_16x16x32_f16(false, af0, false, bf,
                                                  (short)0, acc0, false, false);
    acc1 = __builtin_amdgcn_wmma_f32_16x16x32_f16(false, af1, false, bf,
                                                  (short)0, acc1, false, false);
    __builtin_amdgcn_wave_barrier();
  }

  // redistribute C fragments -> per-pixel offset columns (wave-private LDS)
#pragma unroll
  for (int v = 0; v < 8; ++v) {
    const int r0 = v + 8 * hsel;
    myO[r0 * 16 + p] = acc0[v];
    const int r1 = 16 + r0;
    if (r1 < OFFC) myO[r1 * 16 + p] = acc1[v];
  }
  asm volatile("s_wait_dscnt 0" ::: "memory");
  __builtin_amdgcn_wave_barrier();

  // ------------------- Phase B: deformable einsum via WMMA -------------------
  acc0 = (v8f){}; acc1 = (v8f){};

  for (int kk = 0; kk < KK; ++kk) {
    const int ky = kk / 3, kx = kk % 3;
    const float dy = myO[(2 * kk    ) * 16 + p];
    const float dx = myO[(2 * kk + 1) * 16 + p];
    const float py = (float)(oy - PAD_ + ky) + dy;
    const float px = (float)(ox - PAD_ + kx) + dx;
    const float fy = floorf(py), fx = floorf(px);
    const float wy1 = py - fy, wx1 = px - fx;
    const float wy0 = 1.0f - wy1, wx0 = 1.0f - wx1;
    const int iy0 = (int)fy, ix0 = (int)fx;
    const int iy1 = iy0 + 1, ix1 = ix0 + 1;
    const bool vy0 = (iy0 >= 0) & (iy0 < H_), vy1 = (iy1 >= 0) & (iy1 < H_);
    const bool vx0 = (ix0 >= 0) & (ix0 < W_), vx1 = (ix1 >= 0) & (ix1 < W_);
    const float w00 = wy0 * wx0 * (float)(vy0 & vx0);
    const float w01 = wy0 * wx1 * (float)(vy0 & vx1);
    const float w10 = wy1 * wx0 * (float)(vy1 & vx0);
    const float w11 = wy1 * wx1 * (float)(vy1 & vx1);
    const int cy0 = min(max(iy0, 0), H_ - 1), cy1 = min(max(iy1, 0), H_ - 1);
    const int cx0 = min(max(ix0, 0), W_ - 1), cx1 = min(max(ix1, 0), W_ - 1);
    const int i00 = cy0 * W_ + cx0, i01 = cy0 * W_ + cx1;
    const int i10 = cy1 * W_ + cx0, i11 = cy1 * W_ + cx1;

    // each lane fills 16 channels of its pixel's B-column (L2-resident gathers)
#pragma unroll
    for (int c = 0; c < 16; ++c) {
      const int cin = hsel * 16 + c;
      const float* xc = xb + cin * HW;
      const float s = w00 * xc[i00] + w01 * xc[i01] + w10 * xc[i10] + w11 * xc[i11];
      myB[p * CIN + cin] = (_Float16)s;
    }
    asm volatile("s_wait_dscnt 0" ::: "memory");
    __builtin_amdgcn_wave_barrier();

    const v8h blo = *(const v8h*)&myB[p * CIN + kbase];
    const v8h bhi = *(const v8h*)&myB[p * CIN + kbase + 16];
    const v16h bf = __builtin_shufflevector(blo, bhi,
        0,1,2,3,4,5,6,7,8,9,10,11,12,13,14,15);

    const _Float16* wr0 = &sWdcn[p     ][kk * CIN + kbase];
    const _Float16* wr1 = &sWdcn[p + 16][kk * CIN + kbase];
    const v8h a0lo = *(const v8h*)(wr0);  const v8h a0hi = *(const v8h*)(wr0 + 16);
    const v8h a1lo = *(const v8h*)(wr1);  const v8h a1hi = *(const v8h*)(wr1 + 16);
    const v16h af0 = __builtin_shufflevector(a0lo, a0hi,
        0,1,2,3,4,5,6,7,8,9,10,11,12,13,14,15);
    const v16h af1 = __builtin_shufflevector(a1lo, a1hi,
        0,1,2,3,4,5,6,7,8,9,10,11,12,13,14,15);

    acc0 = __builtin_amdgcn_wmma_f32_16x16x32_f16(false, af0, false, bf,
                                                  (short)0, acc0, false, false);
    acc1 = __builtin_amdgcn_wmma_f32_16x16x32_f16(false, af1, false, bf,
                                                  (short)0, acc1, false, false);
    __builtin_amdgcn_wave_barrier();
  }

  // C/D layout: lanes 0-15 -> M = v, lanes 16-31 -> M = v+8 ; N = p. ReLU+store.
#pragma unroll
  for (int v = 0; v < 8; ++v) {
    const int m = v + hsel * 8;
    float o0 = acc0[v]; o0 = o0 > 0.0f ? o0 : 0.0f;
    float o1 = acc1[v]; o1 = o1 > 0.0f ? o1 : 0.0f;
    out[((size_t)(b * COUT + m     )) * HW + rem] = o0;
    out[((size_t)(b * COUT + m + 16)) * HW + rem] = o1;
  }
}

// ---------------------------------------------------------------------------
extern "C" void kernel_launch(void* const* d_in, const int* in_sizes, int n_in,
                              void* d_out, int out_size, void* d_ws, size_t ws_size,
                              hipStream_t stream) {
  const float* x     = (const float*)d_in[0];
  const float* w_off = (const float*)d_in[1];
  const float* b_off = (const float*)d_in[2];
  const float* w_dcn = (const float*)d_in[3];
  float*       out   = (float*)d_out;
  (void)d_ws; (void)ws_size; (void)in_sizes; (void)n_in; (void)out_size;

  const int npix   = B_ * HW;          // 204800
  const int ntiles = npix / 16;        // 12800, exact
  dcn_fused_wmma_kernel<<<ntiles / WAVES, WAVES * 32, 0, stream>>>(
      x, w_off, b_off, w_dcn, out);
}
